// SA_56410100466105
// MI455X (gfx1250) — matrix operand order repaired and checked by reading, not compiled, not run on hardware
//
#include <hip/hip_runtime.h>

// ---------------------------------------------------------------------------
// CDNA5 / gfx1250 fused implementation. All GEMMs use v_wmma_f32_16x16x32_bf16
// (wave32, 16x16 tiles, f32 accumulation). Weights pre-packed into WMMA
// B-fragment order (32 contiguous bytes per lane). GEMM kernels register-block
// 4 N-tiles per wave so each A-fragment load feeds 4 WMMAs.
// ---------------------------------------------------------------------------

#define BATCH 8
#define CF    512
#define CG    256
#define HIDD  256
#define HW    4096
#define WD    64
#define COUT  256

typedef __attribute__((ext_vector_type(16))) __bf16 v16bf;
typedef __attribute__((ext_vector_type(8)))  float  v8f;

// ---- workspace layout (bytes) ---------------------------------------------
#define OFF_WQ   ((size_t)0)              // 512*256*2   = 262144
#define OFF_WK   ((size_t)262144)         // 256*256*2   = 131072
#define OFF_WV   ((size_t)393216)
#define OFF_WO   ((size_t)524288)
#define OFF_EFP  ((size_t)655360)         // 512*256*2   = 262144
#define OFF_OUTC ((size_t)917504)         // 131072
#define OFF_XF   ((size_t)1048576)        // 8*4096*512*2 = 33554432  (aliased: a_i f32 later)
#define OFF_XRAW (OFF_XF   + (size_t)33554432)   // 33554432
#define OFF_XG   (OFF_XRAW + (size_t)33554432)   // 8*4096*256*2 = 16777216 (aliased: e_a later)
#define OFF_QF   (OFF_XG   + (size_t)16777216)   // 8*4096*256*4 = 33554432
#define OFF_QB   (OFF_QF   + (size_t)33554432)   // 16777216 (aliased: combined later)
#define OFF_KP   (OFF_QB   + (size_t)16777216)   // 8*16*256*2 = 65536
#define OFF_VP   (OFF_KP   + (size_t)65536)      // 8*256*32*2 = 131072
#define OFF_AI   OFF_XF
#define OFF_COMB OFF_QB
#define OFF_EA   OFF_XG

// ---- helpers ---------------------------------------------------------------
__device__ __forceinline__ unsigned short f2bf(float f) {
    unsigned u = __float_as_uint(f);
    unsigned r = (u + 0x7FFFu + ((u >> 16) & 1u)) >> 16;
    return (unsigned short)r;
}

__device__ __forceinline__ v16bf ld_frag16(const unsigned short* p) {
    union { v16bf v; uint4 q[2]; } t;
    t.q[0] = *reinterpret_cast<const uint4*>(p);
    t.q[1] = *reinterpret_cast<const uint4*>(p + 8);
    return t.v;
}

// A-fragment (16x32 bf16, M x K): lane<16 holds K [k0..k0+15],
// lane>=16 holds [k0+16..k0+31]  (kb = (lane>>4)*8)
__device__ __forceinline__ v16bf ld_afrag(const unsigned short* row, int k0, int kb) {
    union { v16bf v; uint4 q[2]; } t;
    t.q[0] = *reinterpret_cast<const uint4*>(row + k0 + 2 * kb);
    t.q[1] = *reinterpret_cast<const uint4*>(row + k0 + 2 * kb + 8);
    return t.v;
}

__device__ __forceinline__ v8f wmma_bf(v16bf a, v16bf b, v8f c) {
    return __builtin_amdgcn_wmma_f32_16x16x32_bf16(false, a, false, b, (short)0, c,
                                                   false, false);
}

// ---------------------------------------------------------------------------
// K0: pack an f32 weight matrix [K][N] (or [N][K] if trans) into WMMA
// B-fragment order: dst[(kt*NT+nt)*512 + lane*16 + h],
//   K = kt*32 + (lane>=16?16:0) + h ,  N = nt*16 + (lane&15)
// ---------------------------------------------------------------------------
__global__ void pack_b_kernel(const float* __restrict__ src,
                              unsigned short* __restrict__ dst,
                              int Kdim, int Ncols, int trans) {
    int idx = blockIdx.x * 256 + threadIdx.x;
    if (idx >= Kdim * Ncols) return;
    int h    = idx & 15;
    int lane = (idx >> 4) & 31;
    int frag = idx >> 9;
    int NT = Ncols >> 4;
    int kt = frag / NT, nt = frag % NT;
    int K = kt * 32 + ((lane >> 4) << 4) + h;
    int N = nt * 16 + (lane & 15);
    float v = trans ? src[(size_t)N * Kdim + K] : src[(size_t)K * Ncols + N];
    dst[idx] = f2bf(v);
}

// ---------------------------------------------------------------------------
// K1: LayerNorm over 512 channels of e_f, one wave per pixel.
// Writes normalized bf16 (xf, pixel-major) and raw bf16 transpose (xraw).
// ---------------------------------------------------------------------------
__global__ __launch_bounds__(256) void lnf_kernel(
    const float* __restrict__ ef, const float* __restrict__ g,
    const float* __restrict__ bta, unsigned short* __restrict__ xf,
    unsigned short* __restrict__ xraw) {
    int wid  = blockIdx.x * 8 + (threadIdx.x >> 5);
    int lane = threadIdx.x & 31;
    int b = wid >> 12, p = wid & 4095;
    const float* base = ef + ((size_t)b * CF) * HW + p;
    float vals[16], s = 0.f, s2 = 0.f;
#pragma unroll
    for (int i = 0; i < 16; ++i) {
        float v = base[(size_t)(lane + 32 * i) * HW];
        vals[i] = v; s += v; s2 += v * v;
    }
#pragma unroll
    for (int m = 16; m >= 1; m >>= 1) {
        s  += __shfl_xor(s,  m, 32);
        s2 += __shfl_xor(s2, m, 32);
    }
    float mean = s * (1.f / CF);
    float var  = s2 * (1.f / CF) - mean * mean;
    float rstd = rsqrtf(var + 1e-5f);
    size_t ro = ((size_t)b * HW + p) * CF;
#pragma unroll
    for (int i = 0; i < 16; ++i) {
        int c = lane + 32 * i;
        xraw[ro + c] = f2bf(vals[i]);
        xf[ro + c]   = f2bf((vals[i] - mean) * rstd * g[c] + bta[c]);
    }
}

// K2: LayerNorm over 256 channels of e_g -> xg (bf16, pixel-major)
__global__ __launch_bounds__(256) void lng_kernel(
    const float* __restrict__ eg, const float* __restrict__ g,
    const float* __restrict__ bta, unsigned short* __restrict__ xg) {
    int wid  = blockIdx.x * 8 + (threadIdx.x >> 5);
    int lane = threadIdx.x & 31;
    int b = wid >> 12, p = wid & 4095;
    const float* base = eg + ((size_t)b * CG) * HW + p;
    float vals[8], s = 0.f, s2 = 0.f;
#pragma unroll
    for (int i = 0; i < 8; ++i) {
        float v = base[(size_t)(lane + 32 * i) * HW];
        vals[i] = v; s += v; s2 += v * v;
    }
#pragma unroll
    for (int m = 16; m >= 1; m >>= 1) {
        s  += __shfl_xor(s,  m, 32);
        s2 += __shfl_xor(s2, m, 32);
    }
    float mean = s * (1.f / CG);
    float var  = s2 * (1.f / CG) - mean * mean;
    float rstd = rsqrtf(var + 1e-5f);
    size_t ro = ((size_t)b * HW + p) * CG;
#pragma unroll
    for (int i = 0; i < 8; ++i) {
        int c = lane + 32 * i;
        xg[ro + c] = f2bf((vals[i] - mean) * rstd * g[c] + bta[c]);
    }
}

// ---------------------------------------------------------------------------
// K3: Q = xf @ wq + bq   (M=pixels, K=512, N=256). One wave per 16(M)x64(N)
// block: 4 N-tiles share each A-fragment load (1 A + 4 B loads -> 4 WMMAs).
// Stores q as f32 (depthwise conv input) and bf16 (attention A input).
// ---------------------------------------------------------------------------
__global__ __launch_bounds__(128) void qproj_kernel(
    const unsigned short* __restrict__ xf, const unsigned short* __restrict__ wqp,
    const float* __restrict__ bq, float* __restrict__ qf,
    unsigned short* __restrict__ qb) {
    int wv = threadIdx.x >> 5, lane = threadIdx.x & 31;
    int b = blockIdx.x >> 8, mt = blockIdx.x & 255;
    int nt0 = wv * 4;
    int row = lane & 15, kb = (lane >> 4) * 8;
    const unsigned short* arow = xf + ((size_t)b * HW + mt * 16 + row) * CF;
    v8f acc[4];
#pragma unroll
    for (int t = 0; t < 4; ++t) acc[t] = (v8f){0.f,0.f,0.f,0.f,0.f,0.f,0.f,0.f};
#pragma unroll
    for (int kt = 0; kt < 16; ++kt) {
        v16bf A = ld_afrag(arow, kt * 32, kb);
#pragma unroll
        for (int t = 0; t < 4; ++t) {
            v16bf B = ld_frag16(wqp + (((size_t)kt * 16 + nt0 + t) << 9) + lane * 16);
            acc[t] = wmma_bf(A, B, acc[t]);
        }
    }
    int n = lane & 15, mb = (lane >> 4) * 8;
#pragma unroll
    for (int t = 0; t < 4; ++t) {
        float bias = bq[(nt0 + t) * 16 + n];
#pragma unroll
        for (int r = 0; r < 8; ++r) {
            float v = acc[t][r] + bias;
            size_t o = ((size_t)b * HW + mt * 16 + mb + r) * HIDD + (nt0 + t) * 16 + n;
            qf[o] = v;
            qb[o] = f2bf(v);
        }
    }
}

// ---------------------------------------------------------------------------
// K4: K/V projections fused with 16x16 max/avg pooling. One block per pooled
// cell (256 pixels). kpool: [b][cell][hid]; vpoolT: [b][hid][32] (K-padded 0).
// ---------------------------------------------------------------------------
__global__ __launch_bounds__(256) void kvpool_kernel(
    const unsigned short* __restrict__ xg, const unsigned short* __restrict__ wkp,
    const unsigned short* __restrict__ wvp, const float* __restrict__ bk,
    const float* __restrict__ bv, unsigned short* __restrict__ kpool,
    unsigned short* __restrict__ vpoolT) {
    int b = blockIdx.x >> 4, cell = blockIdx.x & 15;
    int cy = cell >> 2, cx = cell & 3;
    int wv = threadIdx.x >> 5, lane = threadIdx.x & 31;
    int row = lane & 15, kb = (lane >> 4) * 8;
#pragma unroll
    for (int which = 0; which < 2; ++which) {
        int nt = wv + which * 8;
        float tmax = -3.4e38f, tsum = 0.f;
        for (int rt = 0; rt < 16; ++rt) {
            int p0 = (cy * 16 + rt) * WD + cx * 16;
            const unsigned short* arow = xg + ((size_t)b * HW + p0 + row) * CG;
            v8f ack = {0.f, 0.f, 0.f, 0.f, 0.f, 0.f, 0.f, 0.f};
            v8f acv = {0.f, 0.f, 0.f, 0.f, 0.f, 0.f, 0.f, 0.f};
#pragma unroll
            for (int kt = 0; kt < 8; ++kt) {
                v16bf A  = ld_afrag(arow, kt * 32, kb);
                v16bf Bk = ld_frag16(wkp + (((size_t)kt * 16 + nt) << 9) + lane * 16);
                v16bf Bv = ld_frag16(wvp + (((size_t)kt * 16 + nt) << 9) + lane * 16);
                ack = wmma_bf(A, Bk, ack);
                acv = wmma_bf(A, Bv, acv);
            }
            float mx = ack[0], sm = acv[0];
#pragma unroll
            for (int r = 1; r < 8; ++r) { mx = fmaxf(mx, ack[r]); sm += acv[r]; }
            mx = fmaxf(mx, __shfl_xor(mx, 16, 32));   // join the two M-halves
            sm += __shfl_xor(sm, 16, 32);
            tmax = fmaxf(tmax, mx);
            tsum += sm;
        }
        int n = nt * 16 + (lane & 15);
        if (lane < 16) {
            kpool[((size_t)b * 16 + cell) * HIDD + n]  = f2bf(tmax + bk[n]);
            vpoolT[((size_t)b * HIDD + n) * 32 + cell] = f2bf(tsum * (1.f / 256.f) + bv[n]);
            if (cell == 0) {
#pragma unroll
                for (int i = 0; i < 16; ++i)
                    vpoolT[((size_t)b * HIDD + n) * 32 + 16 + i] = 0;
            }
        }
    }
}

// ---------------------------------------------------------------------------
// K5: attention. scores = q @ k^T (K=256), in-register softmax over 16 keys,
// LDS transpose -> A-fragment (K padded to 32 with zeros), a_i = attn @ v.
// ---------------------------------------------------------------------------
__global__ __launch_bounds__(128) void attn_kernel(
    const unsigned short* __restrict__ qb, const unsigned short* __restrict__ kpool,
    const unsigned short* __restrict__ vpoolT, float* __restrict__ ai) {
    __shared__ unsigned short att[4][16][16];
    int wv = threadIdx.x >> 5, lane = threadIdx.x & 31;
    int gw = blockIdx.x * 4 + wv;
    int b = gw >> 8, mt = gw & 255;
    int row = lane & 15, kb = (lane >> 4) * 8;

    const unsigned short* arow = qb + ((size_t)b * HW + mt * 16 + row) * HIDD;
    const unsigned short* krow = kpool + ((size_t)b * 16 + (lane & 15)) * HIDD +
                                 ((lane >> 4) << 4);
    v8f s = {0.f, 0.f, 0.f, 0.f, 0.f, 0.f, 0.f, 0.f};
#pragma unroll
    for (int kt = 0; kt < 8; ++kt) {
        v16bf A = ld_afrag(arow, kt * 32, kb);
        v16bf B = ld_frag16(krow + kt * 32);   // B[k=hid][n=cell] from kpool row
        s = wmma_bf(A, B, s);
    }
    // softmax per M-row (values for row live across the 16 lanes of a half)
    const float scale = 0.0625f;  // 1/sqrt(256)
#pragma unroll
    for (int r = 0; r < 8; ++r) {
        float v = s[r] * scale;
        float mx = v;
#pragma unroll
        for (int m = 1; m < 16; m <<= 1) mx = fmaxf(mx, __shfl_xor(mx, m, 32));
        float ex = expf(v - mx);
        float sum = ex;
#pragma unroll
        for (int m = 1; m < 16; m <<= 1) sum += __shfl_xor(sum, m, 32);
        att[wv][((lane >> 4) << 3) + r][lane & 15] = f2bf(ex / sum);
    }
    __syncthreads();
    // rebuild A-fragment of attn (16x32, K>=16 are zeros)
    union { v16bf v; unsigned short u[16]; } ta;
#pragma unroll
    for (int h = 0; h < 16; ++h) {
        int k = 2 * kb + h;
        ta.u[h] = (k < 16) ? att[wv][row][k] : (unsigned short)0;
    }
#pragma unroll
    for (int nt = 0; nt < 16; ++nt) {
        const unsigned short* vrow =
            vpoolT + ((size_t)b * HIDD + nt * 16 + (lane & 15)) * 32 +
            ((lane >> 4) << 4);
        v8f c = {0.f, 0.f, 0.f, 0.f, 0.f, 0.f, 0.f, 0.f};
        c = wmma_bf(ta.v, ld_frag16(vrow), c);
        int n = lane & 15, mb = (lane >> 4) * 8;
#pragma unroll
        for (int r = 0; r < 8; ++r)
            ai[((size_t)b * HW + mt * 16 + mb + r) * HIDD + nt * 16 + n] = c[r];
    }
}

// ---------------------------------------------------------------------------
// K6: depthwise 3x3 conv on q (zero pad) + a_i -> combined (bf16)
// ---------------------------------------------------------------------------
__global__ __launch_bounds__(256) void dwcomb_kernel(
    const float* __restrict__ qf, const float* __restrict__ ai,
    const float* __restrict__ dwk, unsigned short* __restrict__ comb) {
    int idx = blockIdx.x * 256 + threadIdx.x;
    int ch = idx & 255, p = (idx >> 8) & 4095, b = idx >> 20;
    int y = p >> 6, x = p & 63;
    float acc = 0.f;
#pragma unroll
    for (int ky = -1; ky <= 1; ++ky) {
        int yy = y + ky;
        if ((unsigned)yy >= WD) continue;
#pragma unroll
        for (int kx = -1; kx <= 1; ++kx) {
            int xx = x + kx;
            if ((unsigned)xx >= WD) continue;
            acc += qf[((size_t)b * HW + yy * WD + xx) * HIDD + ch] *
                   dwk[ch * 9 + (ky + 1) * 3 + (kx + 1)];
        }
    }
    comb[idx] = f2bf(acc + ai[idx]);
}

// ---------------------------------------------------------------------------
// K7: comb = combined @ wo + bo (K=256) and efp = e_f_raw @ efp_w^T + efp_b
// (K=512), gated: e_a = efp * comb -> bf16. 4 N-tiles per wave share A loads.
// ---------------------------------------------------------------------------
__global__ __launch_bounds__(128) void gate_kernel(
    const unsigned short* __restrict__ comb, const unsigned short* __restrict__ wop,
    const float* __restrict__ bo, const unsigned short* __restrict__ xraw,
    const unsigned short* __restrict__ efpp, const float* __restrict__ efpb,
    unsigned short* __restrict__ ea) {
    int wv = threadIdx.x >> 5, lane = threadIdx.x & 31;
    int b = blockIdx.x >> 8, mt = blockIdx.x & 255;
    int nt0 = wv * 4;
    int row = lane & 15, kb = (lane >> 4) * 8;
    const unsigned short* crow = comb + ((size_t)b * HW + mt * 16 + row) * HIDD;
    const unsigned short* rrow = xraw + ((size_t)b * HW + mt * 16 + row) * CF;
    v8f a1[4], a2[4];
#pragma unroll
    for (int t = 0; t < 4; ++t) {
        a1[t] = (v8f){0.f,0.f,0.f,0.f,0.f,0.f,0.f,0.f};
        a2[t] = (v8f){0.f,0.f,0.f,0.f,0.f,0.f,0.f,0.f};
    }
#pragma unroll
    for (int kt = 0; kt < 8; ++kt) {
        v16bf A = ld_afrag(crow, kt * 32, kb);
#pragma unroll
        for (int t = 0; t < 4; ++t) {
            v16bf B = ld_frag16(wop + (((size_t)kt * 16 + nt0 + t) << 9) + lane * 16);
            a1[t] = wmma_bf(A, B, a1[t]);
        }
    }
#pragma unroll
    for (int kt = 0; kt < 16; ++kt) {
        v16bf A = ld_afrag(rrow, kt * 32, kb);
#pragma unroll
        for (int t = 0; t < 4; ++t) {
            v16bf B = ld_frag16(efpp + (((size_t)kt * 16 + nt0 + t) << 9) + lane * 16);
            a2[t] = wmma_bf(A, B, a2[t]);
        }
    }
    int n = lane & 15, mb = (lane >> 4) * 8;
#pragma unroll
    for (int t = 0; t < 4; ++t) {
        int nn = (nt0 + t) * 16 + n;
        float b1 = bo[nn], b2 = efpb[nn];
#pragma unroll
        for (int r = 0; r < 8; ++r) {
            float cv = a1[t][r] + b1;
            float ev = a2[t][r] + b2;
            ea[((size_t)b * HW + mt * 16 + mb + r) * HIDD + nn] = f2bf(cv * ev);
        }
    }
}

// ---------------------------------------------------------------------------
// K8: out = ReLU(BN(e_a @ outc_w^T + outc_b)), NCHW store via LDS transpose.
// 4 N-tiles (64 output channels) per wave share A loads.
// ---------------------------------------------------------------------------
__global__ __launch_bounds__(128) void out_kernel(
    const unsigned short* __restrict__ ea, const unsigned short* __restrict__ outcp,
    const float* __restrict__ ocb, const float* __restrict__ bng,
    const float* __restrict__ bnb, const float* __restrict__ bnm,
    const float* __restrict__ bnv, float* __restrict__ out) {
    __shared__ float tile[4][16][68];
    int wv = threadIdx.x >> 5, lane = threadIdx.x & 31;
    int b = blockIdx.x >> 8, mt = blockIdx.x & 255;
    int nt0 = wv * 4;
    int row = lane & 15, kb = (lane >> 4) * 8;
    const unsigned short* arow = ea + ((size_t)b * HW + mt * 16 + row) * HIDD;
    v8f acc[4];
#pragma unroll
    for (int t = 0; t < 4; ++t) acc[t] = (v8f){0.f,0.f,0.f,0.f,0.f,0.f,0.f,0.f};
#pragma unroll
    for (int kt = 0; kt < 8; ++kt) {
        v16bf A = ld_afrag(arow, kt * 32, kb);
#pragma unroll
        for (int t = 0; t < 4; ++t) {
            v16bf B = ld_frag16(outcp + (((size_t)kt * 16 + nt0 + t) << 9) + lane * 16);
            acc[t] = wmma_bf(A, B, acc[t]);
        }
    }
    int n = lane & 15, mb = (lane >> 4) * 8;
#pragma unroll
    for (int t = 0; t < 4; ++t) {
        int o = (nt0 + t) * 16 + n;
        float sc = bng[o] * rsqrtf(bnv[o] + 1e-5f);
        float sh = bnb[o] - bnm[o] * sc + ocb[o] * sc;
#pragma unroll
        for (int r = 0; r < 8; ++r) {
            float v = acc[t][r] * sc + sh;
            tile[wv][mb + r][t * 16 + n] = v > 0.f ? v : 0.f;
        }
    }
    __syncthreads();
    int m2 = lane & 15, oi = lane >> 4;
#pragma unroll
    for (int t = 0; t < 4; ++t) {
#pragma unroll
        for (int i = 0; i < 8; ++i) {
            int ol = t * 16 + oi + 2 * i;
            out[((size_t)b * COUT + nt0 * 16 + ol) * HW + mt * 16 + m2] =
                tile[wv][m2][ol];
        }
    }
}

// ---------------------------------------------------------------------------
extern "C" void kernel_launch(void* const* d_in, const int* in_sizes, int n_in,
                              void* d_out, int out_size, void* d_ws, size_t ws_size,
                              hipStream_t stream) {
    (void)in_sizes; (void)n_in; (void)out_size; (void)ws_size;
    const float* e_f   = (const float*)d_in[0];
    const float* e_g   = (const float*)d_in[1];
    // d_in[2] l_feat, d_in[3] l_mask: unused by the reference computation
    const float* ln_f_g = (const float*)d_in[4];
    const float* ln_f_b = (const float*)d_in[5];
    const float* ln_g_g = (const float*)d_in[6];
    const float* ln_g_b = (const float*)d_in[7];
    const float* wq  = (const float*)d_in[8];
    const float* bq  = (const float*)d_in[9];
    const float* wk  = (const float*)d_in[10];
    const float* bk  = (const float*)d_in[11];
    const float* wvw = (const float*)d_in[12];
    const float* bv  = (const float*)d_in[13];
    const float* wo  = (const float*)d_in[14];
    const float* bo  = (const float*)d_in[15];
    const float* dwk = (const float*)d_in[16];
    const float* efw = (const float*)d_in[17];
    const float* efb = (const float*)d_in[18];
    const float* ocw = (const float*)d_in[19];
    const float* ocb = (const float*)d_in[20];
    const float* bng = (const float*)d_in[21];
    const float* bnb = (const float*)d_in[22];
    const float* bnm = (const float*)d_in[23];
    const float* bnv = (const float*)d_in[24];
    float* outp = (float*)d_out;

    char* ws = (char*)d_ws;
    unsigned short* wqp   = (unsigned short*)(ws + OFF_WQ);
    unsigned short* wkp   = (unsigned short*)(ws + OFF_WK);
    unsigned short* wvp   = (unsigned short*)(ws + OFF_WV);
    unsigned short* wop   = (unsigned short*)(ws + OFF_WO);
    unsigned short* efpp  = (unsigned short*)(ws + OFF_EFP);
    unsigned short* outcp = (unsigned short*)(ws + OFF_OUTC);
    unsigned short* xf    = (unsigned short*)(ws + OFF_XF);
    unsigned short* xraw  = (unsigned short*)(ws + OFF_XRAW);
    unsigned short* xg    = (unsigned short*)(ws + OFF_XG);
    float*          qf    = (float*)(ws + OFF_QF);
    unsigned short* qb    = (unsigned short*)(ws + OFF_QB);
    unsigned short* kpool = (unsigned short*)(ws + OFF_KP);
    unsigned short* vpoolT= (unsigned short*)(ws + OFF_VP);
    float*          ai    = (float*)(ws + OFF_AI);    // aliases xf (dead after K3)
    unsigned short* comb  = (unsigned short*)(ws + OFF_COMB); // aliases qb
    unsigned short* ea    = (unsigned short*)(ws + OFF_EA);   // aliases xg

    // K0: weight packing (all tiny; ~1 MB total -> resident in L2)
    pack_b_kernel<<<(512 * 256 + 255) / 256, 256, 0, stream>>>(wq,  wqp,   512, 256, 0);
    pack_b_kernel<<<(256 * 256 + 255) / 256, 256, 0, stream>>>(wk,  wkp,   256, 256, 0);
    pack_b_kernel<<<(256 * 256 + 255) / 256, 256, 0, stream>>>(wvw, wvp,   256, 256, 0);
    pack_b_kernel<<<(256 * 256 + 255) / 256, 256, 0, stream>>>(wo,  wop,   256, 256, 0);
    pack_b_kernel<<<(512 * 256 + 255) / 256, 256, 0, stream>>>(efw, efpp,  512, 256, 1);
    pack_b_kernel<<<(256 * 256 + 255) / 256, 256, 0, stream>>>(ocw, outcp, 256, 256, 1);

    // K1/K2: layernorms (wave per pixel)
    lnf_kernel<<<BATCH * HW / 8, 256, 0, stream>>>(e_f, ln_f_g, ln_f_b, xf, xraw);
    lng_kernel<<<BATCH * HW / 8, 256, 0, stream>>>(e_g, ln_g_g, ln_g_b, xg);

    // K3: Q projection (WMMA, 4 N-tiles/wave)
    qproj_kernel<<<BATCH * 256, 128, 0, stream>>>(xf, wqp, bq, qf, qb);

    // K4: K/V projections + 16x16 pooling (WMMA + shuffle reductions)
    kvpool_kernel<<<BATCH * 16, 256, 0, stream>>>(xg, wkp, wvp, bk, bv, kpool, vpoolT);

    // K5: pooled attention (WMMA QK^T, softmax, WMMA attn*V)
    attn_kernel<<<BATCH * 256 / 4, 128, 0, stream>>>(qb, kpool, vpoolT, ai);

    // K6: depthwise 3x3 + a_i
    dwcomb_kernel<<<BATCH * HW * HIDD / 256, 256, 0, stream>>>(qf, ai, dwk, comb);

    // K7: (comb@wo+bo) * (e_f@efp^T+efp_b)  (WMMA, 4 N-tiles/wave)
    gate_kernel<<<BATCH * 256, 128, 0, stream>>>(comb, wop, bo, xraw, efpp, efb, ea);

    // K8: output projection + BN + ReLU, NCHW store (WMMA, 4 N-tiles/wave)
    out_kernel<<<BATCH * 256, 128, 0, stream>>>(ea, outcp, ocb, bng, bnb, bnm, bnv,
                                                outp);
}
